// WindowAttention3D_bac_65927747994176
// MI455X (gfx1250) — matrix operand order repaired and checked by reading, not compiled, not run on hardware
//
#include <hip/hip_runtime.h>
#include <math.h>

// ---------------------------------------------------------------------------
// WindowAttention3D fused kernels for gfx1250 (MI455X), wave32 + WMMA bf16.
//   Kernel 1: QKV projection GEMM (bf16 WMMA, f32 accum) -> bf16 q/k/v in ws
//   Kernel 2: per-(window,head) fused attention: S=QK^T (WMMA), bias+mask+
//             softmax in LDS, O=PV (WMMA), direct [B,N,C] f32 output.
// All WMMA fragments are loaded as per-lane-contiguous 16B chunks
// (ds_load_b128) by construction of the LDS layouts.
// ---------------------------------------------------------------------------

typedef __attribute__((ext_vector_type(16))) __bf16 v16bf;
typedef __attribute__((ext_vector_type(8)))  float  v8f;

#define B_WIN   512
#define NTOK    98
#define CDIM    128
#define NHEADS  4
#define HD      32
#define NWIN    256
#define QKV_ELEMS ((long)B_WIN * NHEADS * NTOK * HD)   // per q/k/v, bf16

union Frag16 { v16bf f; uint4 q[2]; };

__device__ __forceinline__ v8f zero8() {
    v8f z;
#pragma unroll
    for (int i = 0; i < 8; ++i) z[i] = 0.0f;
    return z;
}

__device__ __forceinline__ v16bf zero16() {
    Frag16 r;
    r.q[0] = make_uint4(0u, 0u, 0u, 0u);
    r.q[1] = make_uint4(0u, 0u, 0u, 0u);
    return r.f;
}

// A-matrix 16x32 bf16 fragment (ISA 7.12.2): lane L -> row M = L&15,
// half = L>>4. Lane data = two contiguous 16B runs of the source row:
//   elements [k0 + half*8 .. +7]  and  [k0 + 16 + half*8 .. +7].
// Requires ld (elements) multiple of 8 and k0 multiple of 8.
__device__ __forceinline__ v16bf load_frag_a(const __bf16* m, int ld,
                                             int row0, int nrows, int k0) {
    unsigned lane = threadIdx.x & 31u;
    int row  = row0 + (int)(lane & 15u);
    int half = (int)(lane >> 4) & 1;
    if (row >= nrows) return zero16();
    const __bf16* p = m + (long)row * ld + k0 + half * 8;
    Frag16 r;
    r.q[0] = *(const uint4*)(p);
    r.q[1] = *(const uint4*)(p + 16);
    return r.f;
}

// B-matrix 32x16 bf16 fragment where B[k][n] = src[(n0+n)*ld + k0 + k]
// (i.e. the B operand stored transposed/row-major per output column).
// Lane data = two contiguous 16B runs: elements [k0 + half*8 .. +7] and
// [k0 + 16 + half*8 .. +7].
__device__ __forceinline__ v16bf load_frag_bT(const __bf16* m, int ld,
                                              int n0, int nrows, int k0) {
    unsigned lane = threadIdx.x & 31u;
    int n    = n0 + (int)(lane & 15u);
    int half = (int)(lane >> 4) & 1;
    if (n >= nrows) return zero16();
    const __bf16* p = m + (long)n * ld + k0 + half * 8;
    Frag16 r;
    r.q[0] = *(const uint4*)(p);
    r.q[1] = *(const uint4*)(p + 16);
    return r.f;
}

__device__ __forceinline__ v8f wmma_bf16(v16bf a, v16bf b, v8f c) {
    // D = A(16x32) * B(32x16) + C, f32 accumulate.
    return __builtin_amdgcn_wmma_f32_16x16x32_bf16(
        false, a, false, b, (short)0, c, false, false);
}

// ---------------------------------------------------------------------------
// Kernel 1: qkv = x @ W^T  (M=50176, N=384, K=128), epilogue scatters into
// bf16 q/k/v workspaces laid out [B_, H, N, hd]; q pre-scaled by hd^-0.5.
// Grid: (784, 6) blocks of 128 threads; each block does a 64x64 tile.
// ---------------------------------------------------------------------------
__global__ __launch_bounds__(128) void qkv_gemm_kernel(
    const float* __restrict__ x, const float* __restrict__ w,
    __bf16* __restrict__ qws, __bf16* __restrict__ kws,
    __bf16* __restrict__ vws) {
    __shared__ __align__(16) __bf16 As[64 * 128];   // 16 KB, A rows x K
    __shared__ __align__(16) __bf16 Ws[64 * 128];   // 16 KB, W[n][k] row-major

    const int row0 = blockIdx.x * 64;
    const int col0 = blockIdx.y * 64;
    const int tid  = threadIdx.x;

    // Both tiles are contiguous 64x128 f32 regions; load float4, pack 4 bf16
    // (consecutive LDS halves merge into ds_store_b64).
    const float4* xs4 = (const float4*)(x + (long)row0 * 128);
    const float4* ws4 = (const float4*)(w + (long)col0 * 128);
    for (int idx = tid; idx < 2048; idx += 128) {
        float4 t = xs4[idx];
        __bf16* d = As + idx * 4;
        d[0] = (__bf16)t.x; d[1] = (__bf16)t.y;
        d[2] = (__bf16)t.z; d[3] = (__bf16)t.w;
        float4 u = ws4[idx];
        __bf16* e = Ws + idx * 4;
        e[0] = (__bf16)u.x; e[1] = (__bf16)u.y;
        e[2] = (__bf16)u.z; e[3] = (__bf16)u.w;
    }
    __syncthreads();

    const int wv = tid >> 5;                 // wave id: rows wv*16..wv*16+15
    v8f acc[4];
#pragma unroll
    for (int n = 0; n < 4; ++n) acc[n] = zero8();

#pragma unroll
    for (int kc = 0; kc < 4; ++kc) {
        v16bf a = load_frag_a(As, 128, wv * 16, 64, kc * 32);
#pragma unroll
        for (int n = 0; n < 4; ++n) {
            v16bf b = load_frag_bT(Ws, 128, n * 16, 64, kc * 32);
            acc[n] = wmma_bf16(a, b, acc[n]);
        }
    }

    const unsigned lane = tid & 31u;
    const int half = (int)(lane >> 4) & 1;
    const int lnc  = (int)(lane & 15u);
#pragma unroll
    for (int n = 0; n < 4; ++n) {
#pragma unroll
        for (int r = 0; r < 8; ++r) {
            int i = row0 + wv * 16 + r + 8 * half;      // < 50176 always
            int j = col0 + n * 16 + lnc;                // < 384
            float v = acc[n][r];
            int b = i / NTOK, nn = i % NTOK;
            int s = j >> 7, hh = (j >> 5) & 3, d = j & 31;
            long off = (((long)(b * NHEADS + hh)) * NTOK + nn) * HD + d;
            if (s == 0)      qws[off] = (__bf16)(v * 0.17677669529663687f);
            else if (s == 1) kws[off] = (__bf16)v;
            else             vws[off] = (__bf16)v;
        }
    }
}

// ---------------------------------------------------------------------------
// Kernel 2: fused attention per (window b, head h).
// Grid: (512, 4) blocks of 128 threads (4 waves).
// LDS: Q/K bf16 [98x32] + V^T bf16 [32x128] + S f32 [98x112] = 63.1 KB.
// P (bf16, stride 224 = 448B rows) is overlaid in-place on S's rows.
// ---------------------------------------------------------------------------
__global__ __launch_bounds__(128) void attn_fused_kernel(
    const __bf16* __restrict__ qws, const __bf16* __restrict__ kws,
    const __bf16* __restrict__ vws, const float* __restrict__ bias_table,
    const float* __restrict__ mask, float* __restrict__ out) {
    __shared__ __align__(16) __bf16 Qs[NTOK * HD];   // [token][d]
    __shared__ __align__(16) __bf16 Ks[NTOK * HD];   // [token][d]
    __shared__ __align__(16) __bf16 Vt[HD * 128];    // [d][token], zero-padded
    __shared__ __align__(16) float  Ss[NTOK * 112];
    __bf16* Ps = (__bf16*)Ss;               // overlay, row stride 224 bf16

    const int b   = blockIdx.x;
    const int h   = blockIdx.y;
    const int tid = threadIdx.x;
    const int wv  = tid >> 5;
    const unsigned lane = tid & 31u;
    const int half = (int)(lane >> 4) & 1;
    const int lnc  = (int)(lane & 15u);

    const long base = ((long)(b * NHEADS + h)) * NTOK * HD;
    {   // Q/K: straight uint4 copies (98*32 bf16 = 392 uint4 each)
        const uint4* q4 = (const uint4*)(qws + base);
        const uint4* k4 = (const uint4*)(kws + base);
        uint4* qd = (uint4*)Qs;
        uint4* kd = (uint4*)Ks;
        for (int idx = tid; idx < 392; idx += 128) {
            qd[idx] = q4[idx];
            kd[idx] = k4[idx];
        }
        // V transposed into [d][token] with zero padding for tokens 98..127
        const __bf16* vsrc = vws + base;
        for (int idx = tid; idx < HD * 128; idx += 128) {
            int d = idx >> 7, key = idx & 127;
            Vt[idx] = (key < NTOK) ? vsrc[(long)key * HD + d] : (__bf16)0.0f;
        }
    }
    __syncthreads();

    // ---- S = Q K^T : 7x7 tiles of 16x16, one 16x16x32 WMMA each ----
    for (int t = wv; t < 49; t += 4) {
        int ti = t / 7, tj = t % 7;
        v16bf a  = load_frag_a(Qs, HD, ti * 16, NTOK, 0);
        v16bf bt = load_frag_bT(Ks, HD, tj * 16, NTOK, 0);
        v8f c = wmma_bf16(a, bt, zero8());
#pragma unroll
        for (int r = 0; r < 8; ++r) {
            int row = ti * 16 + r + 8 * half;
            if (row < NTOK) Ss[row * 112 + tj * 16 + lnc] = c[r];
        }
    }
    __syncthreads();

    // ---- bias + mask + softmax, one token row per thread ----
    if (tid < NTOK) {
        const int r  = tid;
        const int zr = r / 49, rr = r - zr * 49;
        const int yr = rr / 7, xr = rr - yr * 7;
        const float* mrow = mask + ((long)(b & (NWIN - 1))) * NTOK * NTOK
                                 + (long)r * NTOK;
        float* srow = Ss + r * 112;
        float mx = -3.4e38f;
        int c2 = 0;
        for (int zc = 0; zc < 2; ++zc) {
            int zb = (zr - zc + 1) * 169;
            for (int yc = 0; yc < 7; ++yc) {
                int yb = zb + (yr - yc + 6) * 13 + xr + 6;   // rpi = yb - xc
#pragma unroll
                for (int xc = 0; xc < 7; ++xc, ++c2) {
                    float t = srow[c2] + bias_table[(yb - xc) * NHEADS + h]
                                       + mrow[c2];
                    srow[c2] = t;
                    mx = fmaxf(mx, t);
                }
            }
        }
        float sum = 0.0f;
        for (int c3 = 0; c3 < NTOK; ++c3) {
            float e = __expf(srow[c3] - mx);
            srow[c3] = e;
            sum += e;
        }
        float inv = 1.0f / sum;
        // In-place bf16 overlay, bf16-pair (b32) stores: at step c3 we write
        // bytes [2*c3, 2*c3+4) after having read through byte 4*c3+7 -> safe.
        unsigned* prow = (unsigned*)((char*)Ss + (long)r * 448);
        for (int c3 = 0; c3 < 128; c3 += 2) {
            float p0 = (c3     < NTOK) ? srow[c3]     * inv : 0.0f;
            float p1 = (c3 + 1 < NTOK) ? srow[c3 + 1] * inv : 0.0f;
            union { __bf16 hh[2]; unsigned u; } pk;
            pk.hh[0] = (__bf16)p0;
            pk.hh[1] = (__bf16)p1;
            prow[c3 >> 1] = pk.u;
        }
    }
    __syncthreads();

    // ---- O = P V : 7 row-tiles x 2 col-tiles, K padded to 128 (P cols
    //      98..127 are exact zeros, V^T tokens >=98 are zero) ----
    for (int t = wv; t < 14; t += 4) {
        int ti = t >> 1, nn = t & 1;
        v8f acc = zero8();
#pragma unroll
        for (int kc = 0; kc < 4; ++kc) {
            v16bf a  = load_frag_a(Ps, 224, ti * 16, NTOK, kc * 32);
            v16bf bb = load_frag_bT(Vt, 128, nn * 16, HD, kc * 32);
            acc = wmma_bf16(a, bb, acc);
        }
#pragma unroll
        for (int r = 0; r < 8; ++r) {
            int row = ti * 16 + r + 8 * half;
            if (row < NTOK)
                out[((long)b * NTOK + row) * CDIM + h * HD + nn * 16 + lnc] =
                    acc[r];
        }
    }
}

extern "C" void kernel_launch(void* const* d_in, const int* in_sizes, int n_in,
                              void* d_out, int out_size, void* d_ws,
                              size_t ws_size, hipStream_t stream) {
    const float* x          = (const float*)d_in[0];
    const float* qkv_w      = (const float*)d_in[1];
    const float* bias_table = (const float*)d_in[2];
    const float* mask       = (const float*)d_in[3];
    float* out = (float*)d_out;

    __bf16* qws = (__bf16*)d_ws;            // 3 * 6.4M bf16 = 36.75 MB scratch
    __bf16* kws = qws + QKV_ELEMS;
    __bf16* vws = kws + QKV_ELEMS;

    dim3 g1(784, 6), blk(128);
    hipLaunchKernelGGL(qkv_gemm_kernel, g1, blk, 0, stream,
                       x, qkv_w, qws, kws, vws);

    dim3 g2(B_WIN, NHEADS);
    hipLaunchKernelGGL(attn_fused_kernel, g2, blk, 0, stream,
                       qws, kws, vws, bias_table, mask, out);
}
